// LocalNodeGATLayer_57140244906495
// MI455X (gfx1250) — matrix operand order repaired and checked by reading, not compiled, not run on hardware
//
#include <hip/hip_runtime.h>
#include <hip/hip_bf16.h>
#include <cstdint>

typedef __attribute__((ext_vector_type(16))) __bf16 v16bf;
typedef __attribute__((ext_vector_type(8)))  float  v8f;

#define D_DIM 128

// ---------------------------------------------------------------------------
// Kernel 1: per-node attention logit  s[n] = W2 . tanh( N[n] @ W1^T )
// GEMM done on bf16 WMMA pipes with 2-term (hi+lo) split for ~fp32 accuracy:
//   A*B ~= Ah*Bh + Al*Bh + Ah*Bl      (3x v_wmma_f32_16x16x32_bf16 per K-tile)
// Block = 256 threads = 8 waves; each wave owns a 16-row tile (128 rows/block).
// W1 staged once per block into LDS as bf16 hi/lo (64 KB) -> aligned ds loads.
// ---------------------------------------------------------------------------
__global__ __launch_bounds__(256)
void gat_score_kernel(const float* __restrict__ Nfeat,
                      const float* __restrict__ W1,
                      const float* __restrict__ W2,
                      float* __restrict__ s,
                      int n_nodes)
{
    __shared__ __bf16 w1hi[D_DIM * D_DIM];   // 32 KB
    __shared__ __bf16 w1lo[D_DIM * D_DIM];   // 32 KB

    // Cooperative stage of W1 as bf16 hi + residual lo.
    for (int idx = threadIdx.x; idx < D_DIM * D_DIM; idx += blockDim.x) {
        float x   = W1[idx];
        __bf16 h  = (__bf16)x;
        w1hi[idx] = h;
        w1lo[idx] = (__bf16)(x - (float)h);
    }
    __syncthreads();

    const int lane = threadIdx.x & 31;        // wave32
    const int wave = threadIdx.x >> 5;
    const int row0 = blockIdx.x * 128 + wave * 16;
    if (row0 >= n_nodes) return;              // no barriers past this point

    const int half = lane >> 4;               // 0: lanes 0-15, 1: lanes 16-31
    const int l16  = lane & 15;
    int arow = row0 + l16;
    if (arow >= n_nodes) arow = n_nodes - 1;  // clamp (harmless dup row)

    // --- Build A fragments (16x32 bf16, contiguous-k layout) for all 4 K-tiles -----
    // lane element i -> K = kt*32 + half*16 + i
    v16bf a_hi[4], a_lo[4];
    const float* nrow = Nfeat + (size_t)arow * D_DIM;
    #pragma unroll
    for (int kt = 0; kt < 4; ++kt) {
        #pragma unroll
        for (int hb = 0; hb < 2; ++hb) {
            const int c0 = kt * 32 + half * 16 + hb * 8;
            #pragma unroll
            for (int j = 0; j < 8; ++j) {
                float  x = nrow[c0 + j];
                __bf16 h = (__bf16)x;
                a_hi[kt][hb * 8 + j] = h;
                a_lo[kt][hb * 8 + j] = (__bf16)(x - (float)h);
            }
        }
    }

    float acc[8];
    #pragma unroll
    for (int v = 0; v < 8; ++v) acc[v] = 0.0f;

    // --- Loop over 8 column tiles of the 128-wide fc1 output --------------
    #pragma unroll
    for (int nt = 0; nt < 8; ++nt) {
        const int n = nt * 16 + l16;          // B column owned by this lane
        const __bf16* bh_base = &w1hi[n * D_DIM];
        const __bf16* bl_base = &w1lo[n * D_DIM];

        v8f c = {};                            // f32 accumulator tile 16x16
        #pragma unroll
        for (int kt = 0; kt < 4; ++kt) {
            // B element i -> K = kt*32 + half*16 + i : 32B-aligned LDS vector
            const int k0 = kt * 32 + half * 16;
            v16bf bh = *(const v16bf*)(bh_base + k0);
            v16bf bl = *(const v16bf*)(bl_base + k0);
            c = __builtin_amdgcn_wmma_f32_16x16x32_bf16(false, a_hi[kt], false, bh,
                                                        (short)0, c, false, false);
            c = __builtin_amdgcn_wmma_f32_16x16x32_bf16(false, a_lo[kt], false, bh,
                                                        (short)0, c, false, false);
            c = __builtin_amdgcn_wmma_f32_16x16x32_bf16(false, a_hi[kt], false, bl,
                                                        (short)0, c, false, false);
        }

        // c[v] holds fc1 output at (row = v + 8*half, col = n); fuse tanh + W2 dot
        const float w2c = W2[n];
        #pragma unroll
        for (int v = 0; v < 8; ++v)
            acc[v] += tanhf(c[v]) * w2c;
    }

    // Reduce across the 16 lanes of each half (each lane covered one column).
    #pragma unroll
    for (int off = 1; off < 16; off <<= 1) {
        #pragma unroll
        for (int v = 0; v < 8; ++v)
            acc[v] += __shfl_xor(acc[v], off, 32);
    }

    if (l16 == 0) {
        #pragma unroll
        for (int v = 0; v < 8; ++v) {
            const int r = row0 + half * 8 + v;
            if (r < n_nodes) s[r] = acc[v];
        }
    }
}

// ---------------------------------------------------------------------------
// Kernel 2: CSR row pointers from the (sorted) dst array via binary search.
// ---------------------------------------------------------------------------
__global__ void gat_rowptr_kernel(const int* __restrict__ dst, int n_edges,
                                  int* __restrict__ rowptr, int n_nodes)
{
    int v = blockIdx.x * blockDim.x + threadIdx.x;
    if (v > n_nodes) return;
    int lo = 0, hi = n_edges;
    while (lo < hi) {                    // lower_bound(dst, v)
        int mid = (lo + hi) >> 1;
        if (dst[mid] < v) lo = mid + 1; else hi = mid;
    }
    rowptr[v] = lo;
}

// ---------------------------------------------------------------------------
// Kernel 3: per-destination softmax + weighted neighbor sum (no atomics;
// edges for each dst are contiguous). Block = 128 threads = one feature dim
// each. N rows are L2-resident (25.6 MB << 192 MB); out uses non-temporal
// stores so the write-once output doesn't evict N from L2.
// ---------------------------------------------------------------------------
__global__ __launch_bounds__(128)
void gat_aggregate_kernel(const float* __restrict__ Nfeat,
                          const int*   __restrict__ src,
                          const float* __restrict__ s,
                          const int*   __restrict__ rowptr,
                          float*       __restrict__ out,
                          int n_nodes)
{
    const int v   = blockIdx.x;
    const int tid = threadIdx.x;           // feature dim
    const int lo  = rowptr[v];
    const int hi  = rowptr[v + 1];
    const int deg = hi - lo;

    float* outp = out + (size_t)v * D_DIM + tid;
    if (deg == 0) {                        // no in-edges -> zeros (matches DGL)
        __builtin_nontemporal_store(0.0f, outp);
        return;
    }

    __shared__ float red[128];
    __shared__ float w_sm[128];
    __shared__ int   i_sm[128];

    // --- segment max (numerically stable softmax) --------------------------
    float lmax = -3.402823466e38f;
    for (int j = lo + tid; j < hi; j += 128)
        lmax = fmaxf(lmax, s[src[j]]);
    red[tid] = lmax;
    __syncthreads();
    for (int st = 64; st > 0; st >>= 1) {
        if (tid < st) red[tid] = fmaxf(red[tid], red[tid + st]);
        __syncthreads();
    }
    const float m = red[0];
    __syncthreads();

    // --- segment sum of exp ------------------------------------------------
    float lsum = 0.0f;
    for (int j = lo + tid; j < hi; j += 128)
        lsum += __expf(s[src[j]] - m);
    red[tid] = lsum;
    __syncthreads();
    for (int st = 64; st > 0; st >>= 1) {
        if (tid < st) red[tid] += red[tid + st];
        __syncthreads();
    }
    const float invz = 1.0f / red[0];
    __syncthreads();

    // --- weighted neighbor accumulation, chunked through LDS ---------------
    float acc = 0.0f;
    for (int base = lo; base < hi; base += 128) {
        const int cnt = min(128, hi - base);
        if (tid < cnt) {
            const int u = src[base + tid];
            i_sm[tid] = u;
            w_sm[tid] = __expf(s[u] - m) * invz;
        }
        __syncthreads();
        for (int jj = 0; jj < cnt; ++jj)
            acc += w_sm[jj] * Nfeat[(size_t)i_sm[jj] * D_DIM + tid];  // coalesced
        __syncthreads();
    }

    __builtin_nontemporal_store(acc, outp);
}

// ---------------------------------------------------------------------------
extern "C" void kernel_launch(void* const* d_in, const int* in_sizes, int n_in,
                              void* d_out, int out_size, void* d_ws, size_t ws_size,
                              hipStream_t stream)
{
    const float* Nfeat = (const float*)d_in[0];   // [n_nodes, 128] f32
    const int*   src   = (const int*)  d_in[1];   // [n_edges]
    const int*   dst   = (const int*)  d_in[2];   // [n_edges] sorted
    const float* W1    = (const float*)d_in[3];   // [128, 128]
    const float* W2    = (const float*)d_in[4];   // [128]
    float*       out   = (float*)d_out;

    const int n_nodes = in_sizes[0] / D_DIM;
    const int n_edges = in_sizes[1];

    // Workspace layout: s[n_nodes] f32, then rowptr[n_nodes+1] i32.
    float* s_buf = (float*)d_ws;
    size_t s_bytes = (((size_t)n_nodes * sizeof(float)) + 255) & ~(size_t)255;
    int* rowptr = (int*)((char*)d_ws + s_bytes);

    // 1) per-node scores (bf16-split WMMA GEMM + tanh + dot)
    dim3 g1((n_nodes + 127) / 128);
    gat_score_kernel<<<g1, 256, 0, stream>>>(Nfeat, W1, W2, s_buf, n_nodes);

    // 2) CSR row pointers from sorted dst
    dim3 g2((n_nodes + 1 + 255) / 256);
    gat_rowptr_kernel<<<g2, 256, 0, stream>>>(dst, n_edges, rowptr, n_nodes);

    // 3) per-node segment softmax + weighted aggregation
    gat_aggregate_kernel<<<n_nodes, 128, 0, stream>>>(Nfeat, src, s_buf,
                                                      rowptr, out, n_nodes);
}
